// PointExtractor_55671366091211
// MI455X (gfx1250) — compile-verified, matrix operand order checked by reading, not run on hardware
//
#include <hip/hip_runtime.h>

typedef __attribute__((ext_vector_type(16))) _Float16     v16h;
typedef __attribute__((ext_vector_type(8)))  float        v8f;
typedef __attribute__((ext_vector_type(4)))  unsigned int v4u;
typedef __attribute__((ext_vector_type(8)))  int          v8i;
typedef __attribute__((ext_vector_type(4)))  int          v4i;
typedef _Float16 f16;

#define P_   64
#define N_   2048
#define M1_  512
#define M2_  128
#define K_   32
#define R1_  0.15f
#define R2_  0.30f
#define INFF 3.402823466e38f

#if defined(__has_builtin)
#if __has_builtin(__builtin_amdgcn_tensor_load_to_lds)
#define USE_TDM 1
#endif
#endif

// ---- d_out layout (floats, concatenated in reference return order) ----
#define OFF_XG     0                        // [P,768]
#define OFF_POSG   (OFF_XG + P_*768)        // [P,3]
#define OFF_BATCHG (OFF_POSG + P_*3)        // [P]
#define OFF_X2     (OFF_BATCHG + P_)        // [P*M2,384]
#define OFF_Q2     (OFF_X2 + P_*M2_*384)    // [P*M2,3]
#define OFF_B2     (OFF_Q2 + P_*M2_*3)      // [P*M2]
#define OFF_VMIN   (OFF_B2 + P_*M2_)        // [P,3]
#define OFF_DIFF   (OFF_VMIN + P_*3)        // [P]

// ---------------- init aux outputs ----------------
__global__ void k_aux(float* out) {
  int i = blockIdx.x * blockDim.x + threadIdx.x;
  if (i < P_ * 768)  out[OFF_XG + i]     = 0.0f;
  if (i < P_ * 3)    out[OFF_POSG + i]   = 0.0f;
  if (i < P_)        out[OFF_BATCHG + i] = (float)i;
  if (i < P_ * M2_)  out[OFF_B2 + i]     = (float)(i / M2_);
}

// ---------------- fp32 weights -> f16, packed fragment-major ----------------
// Packed layout: tile t = kt*(ncols/16)+jt holds 512 halves; lane l owns 16
// contiguous halves at t*512 + l*16, matching the B-operand VGPR layout:
//   col = l&15, kb = (l<16?0:8), half h: i=h>>1, kk=(i<4?kb+2i:16+kb+2(i-4))+(h&1)
__global__ void k_cvt_pack(const float* __restrict__ src, f16* __restrict__ dst,
                           int rows, int ncols, int kpad) {
  int e = blockIdx.x * blockDim.x + threadIdx.x;
  int total = (kpad >> 5) * (ncols >> 4) * 512;
  if (e >= total) return;
  int t = e >> 9, r = e & 511;
  int lane = r >> 4, h = r & 15;
  int ntiles = ncols >> 4;
  int kt = t / ntiles, jt = t % ntiles;
  int cr = lane & 15, kb = (lane >> 4) << 3;
  int i = h >> 1, odd = h & 1;
  int kk = ((i < 4) ? (kb + 2 * i) : (16 + kb + 2 * (i - 4))) + odd;
  int k = kt * 32 + kk, n = jt * 16 + cr;
  dst[e] = (k < rows) ? (f16)src[k * ncols + n] : (f16)0.0f;
}

__device__ __forceinline__ v16h load_b_pk(const f16* Wpk, int ntiles, int kt, int jt, int lane) {
  return *(const v16h*)(Wpk + ((((size_t)kt * ntiles + jt) << 9) + ((size_t)lane << 4)));
}

// A (16x32 f16) from LDS: lane l -> row = l&15, kb = (l<16?0:8)
__device__ __forceinline__ v16h load_a_frag(const f16* base, int stride, int row, int kb) {
  v16h a;
  const f16* rp = base + (size_t)row * stride;
#pragma unroll
  for (int i = 0; i < 8; ++i) {
    int kk = (i < 4) ? (kb + 2 * i) : (16 + kb + 2 * (i - 4));
    a[2 * i]     = rp[kk];
    a[2 * i + 1] = rp[kk + 1];
  }
  return a;
}

// ---------------- per-patch min/max normalize ----------------
__global__ void k_norm(const float* __restrict__ pos, float* __restrict__ pn,
                       float* __restrict__ vminOut, float* __restrict__ diffOut) {
  __shared__ float sred[6][256];
  int p = blockIdx.x, t = threadIdx.x;
  const float* pp = pos + (size_t)p * N_ * 3;
  float mn[3] = {INFF, INFF, INFF}, mx[3] = {-INFF, -INFF, -INFF};
  for (int i = t; i < N_; i += 256)
    for (int d = 0; d < 3; ++d) {
      float v = pp[3 * i + d];
      mn[d] = fminf(mn[d], v); mx[d] = fmaxf(mx[d], v);
    }
  for (int d = 0; d < 3; ++d) { sred[d][t] = mn[d]; sred[3 + d][t] = mx[d]; }
  __syncthreads();
  for (int s = 128; s > 0; s >>= 1) {
    if (t < s)
      for (int d = 0; d < 3; ++d) {
        sred[d][t]     = fminf(sred[d][t],     sred[d][t + s]);
        sred[3 + d][t] = fmaxf(sred[3 + d][t], sred[3 + d][t + s]);
      }
    __syncthreads();
  }
  float m0 = sred[0][0], m1 = sred[1][0], m2 = sred[2][0];
  float dv = fmaxf(fmaxf(sred[3][0] - m0, sred[4][0] - m1), sred[5][0] - m2);
  if (t == 0) {
    vminOut[p * 3 + 0] = m0; vminOut[p * 3 + 1] = m1; vminOut[p * 3 + 2] = m2;
    diffOut[p] = dv;
  }
  float inv = 1.0f / dv;
  for (int i = t; i < N_; i += 256) {
    size_t o = ((size_t)p * N_ + i) * 3;
    pn[o + 0] = (pp[3 * i + 0] - m0) * inv;
    pn[o + 1] = (pp[3 * i + 1] - m1) * inv;
    pn[o + 2] = (pp[3 * i + 2] - m2) * inv;
  }
}

// ---------------- farthest point sampling (one block per patch) ----------------
__global__ void k_fps(const float* __restrict__ pts, int n, int m,
                      int* __restrict__ idx, float* __restrict__ q) {
  extern __shared__ char dynsm[];
  float* dist = (float*)dynsm;
  float* rv   = dist + n;
  int*   ri   = (int*)(rv + 256);
  int p = blockIdx.x, t = threadIdx.x;
  const float* pp = pts + (size_t)p * n * 3;
  float x0 = pp[0], y0 = pp[1], z0 = pp[2];
  for (int i = t; i < n; i += 256) {
    float dx = pp[3 * i] - x0, dy = pp[3 * i + 1] - y0, dz = pp[3 * i + 2] - z0;
    dist[i] = dx * dx + dy * dy + dz * dz;
  }
  if (t == 0) {
    idx[(size_t)p * m] = 0;
    q[(size_t)p * m * 3 + 0] = x0; q[(size_t)p * m * 3 + 1] = y0; q[(size_t)p * m * 3 + 2] = z0;
  }
  __syncthreads();
  for (int it = 1; it < m; ++it) {
    float bv = -1.0f; int bi = 0;
    for (int i = t; i < n; i += 256) {
      float v = dist[i];
      if (v > bv) { bv = v; bi = i; }
    }
    rv[t] = bv; ri[t] = bi;
    __syncthreads();
    for (int s = 128; s > 0; s >>= 1) {
      if (t < s && (rv[t + s] > rv[t] || (rv[t + s] == rv[t] && ri[t + s] < ri[t]))) {
        rv[t] = rv[t + s]; ri[t] = ri[t + s];
      }
      __syncthreads();
    }
    int j = ri[0];
    float jx = pp[3 * j], jy = pp[3 * j + 1], jz = pp[3 * j + 2];
    if (t == 0) {
      idx[(size_t)p * m + it] = j;
      size_t o = ((size_t)p * m + it) * 3;
      q[o] = jx; q[o + 1] = jy; q[o + 2] = jz;
    }
    for (int i = t; i < n; i += 256) {
      float dx = pp[3 * i] - jx, dy = pp[3 * i + 1] - jy, dz = pp[3 * i + 2] - jz;
      dist[i] = fminf(dist[i], dx * dx + dy * dy + dz * dz);
    }
    __syncthreads();
  }
}

// ---------------- K nearest in radius (one wave per center) ----------------
__global__ void k_nbr(const float* __restrict__ pts, const float* __restrict__ q,
                      int n, int m, float r2, int* __restrict__ nidx) {
  extern __shared__ char dynsm[];
  float* d2b = (float*)dynsm;
  int wid = threadIdx.x >> 5, lane = threadIdx.x & 31;
  float* d2 = d2b + (size_t)wid * n;
  int c = blockIdx.x * 8 + wid;
  int p = c / m;
  const float* pp = pts + (size_t)p * n * 3;
  float qx = q[(size_t)c * 3], qy = q[(size_t)c * 3 + 1], qz = q[(size_t)c * 3 + 2];
  for (int i = lane; i < n; i += 32) {
    float dx = pp[3 * i] - qx, dy = pp[3 * i + 1] - qy, dz = pp[3 * i + 2] - qz;
    d2[i] = dx * dx + dy * dy + dz * dz;
  }
  __syncthreads();
  for (int k = 0; k < K_; ++k) {
    float bv = INFF; int bi = 0;
    for (int i = lane; i < n; i += 32) {
      float v = d2[i];
      if (v < bv) { bv = v; bi = i; }
    }
    for (int off = 16; off; off >>= 1) {
      float ov = __shfl_xor(bv, off, 32);
      int   oi = __shfl_xor(bi, off, 32);
      if (ov < bv || (ov == bv && oi < bi)) { bv = ov; bi = oi; }
    }
    int sel = (bv <= r2) ? bi : -1;
    if (lane == 0) {
      nidx[(size_t)c * K_ + k] = sel;
      if (sel >= 0) d2[sel] = INFF;
    }
    __syncthreads();
  }
}

// ---------------- layer1: gather f32 feats -> MLP -> masked max (f16 out) ----------------
__global__ void k_conv(const float* __restrict__ feats, int C,
                       const float* __restrict__ pts, const float* __restrict__ q,
                       const int* __restrict__ nidx,
                       const f16* __restrict__ WaPk, const float* __restrict__ ba, int CPAD, int H1,
                       const f16* __restrict__ WbPk, const float* __restrict__ bb, int H2,
                       f16* __restrict__ out, int n, int m) {
  extern __shared__ char dynsm[];
  int* valid = (int*)dynsm;
  f16* A  = (f16*)(dynsm + 128);
  f16* Hs = A + 32 * CPAD;
  int c = blockIdx.x, lane = threadIdx.x;    // blockDim = 32
  int p = c / m;
  int gi = nidx[(size_t)c * K_ + lane];
  valid[lane] = (gi >= 0);
  int gs = gi < 0 ? 0 : gi;
  float keep = gi >= 0 ? 1.0f : 0.0f;
  float qx = q[(size_t)c * 3], qy = q[(size_t)c * 3 + 1], qz = q[(size_t)c * 3 + 2];
  const float* F  = feats + ((size_t)p * n + gs) * C;
  const float* Pt = pts   + ((size_t)p * n + gs) * 3;
  f16* Ar = A + lane * CPAD;
  for (int j = 0; j < C; ++j) Ar[j] = (f16)(F[j] * keep);
  Ar[C + 0] = (f16)((Pt[0] - qx) * keep);
  Ar[C + 1] = (f16)((Pt[1] - qy) * keep);
  Ar[C + 2] = (f16)((Pt[2] - qz) * keep);
  for (int j = C + 3; j < CPAD; ++j) Ar[j] = (f16)0.0f;
  __builtin_prefetch(WaPk, 0, 1);
  __builtin_prefetch(WbPk, 0, 1);
  __syncthreads();

  int cr = lane & 15;
  int kb = (lane >> 4) << 3;
  for (int mt = 0; mt < 2; ++mt)
    for (int jt = 0; jt < H1 / 16; ++jt) {
      v8f acc = {};
      for (int kt = 0; kt < CPAD / 32; ++kt) {
        v16h a = load_a_frag(A + (size_t)mt * 16 * CPAD + kt * 32, CPAD, cr, kb);
        v16h b = load_b_pk(WaPk, H1 / 16, kt, jt, lane);
        acc = __builtin_amdgcn_wmma_f32_16x16x32_f16(false, a, false, b, (short)0, acc, false, false);
      }
      float bc = ba[jt * 16 + cr];
#pragma unroll
      for (int r = 0; r < 8; ++r) {
        int rr = mt * 16 + ((lane < 16) ? r : 8 + r);
        float v = acc[r] + bc; v = v > 0.0f ? v : 0.0f;
        Hs[(size_t)rr * H1 + jt * 16 + cr] = (f16)v;
      }
    }
  __syncthreads();
  for (int jt = 0; jt < H2 / 16; ++jt) {
    float cmax = -INFF;
    float bc = bb[jt * 16 + cr];
    for (int mt = 0; mt < 2; ++mt) {
      v8f acc = {};
      for (int kt = 0; kt < H1 / 32; ++kt) {
        v16h a = load_a_frag(Hs + (size_t)mt * 16 * H1 + kt * 32, H1, cr, kb);
        v16h b = load_b_pk(WbPk, H2 / 16, kt, jt, lane);
        acc = __builtin_amdgcn_wmma_f32_16x16x32_f16(false, a, false, b, (short)0, acc, false, false);
      }
#pragma unroll
      for (int r = 0; r < 8; ++r) {
        int rr = mt * 16 + ((lane < 16) ? r : 8 + r);
        float v = acc[r] + bc; v = v > 0.0f ? v : 0.0f;
        if (valid[rr]) cmax = fmaxf(cmax, v);
      }
    }
    cmax = fmaxf(cmax, __shfl_xor(cmax, 16, 32));
    if (lane < 16) out[(size_t)c * H2 + jt * 16 + cr] = (f16)cmax;
  }
}

// ---------------- layer2: TDM gather of f16 rows -> MLP -> masked max (f32 out) ----------------
__global__ void k_conv2(const f16* __restrict__ x1h, const float* __restrict__ pts,
                        const float* __restrict__ q, const int* __restrict__ nidx,
                        const f16* __restrict__ WaPk, const float* __restrict__ ba,
                        const f16* __restrict__ WbPk, const float* __restrict__ bb,
                        float* __restrict__ out) {
  const int C = 128, CPAD = 160, H1 = 256, H2 = 384, n = M1_, m = M2_;
  extern __shared__ char dynsm[];
  int* valid = (int*)dynsm;
  f16* A  = (f16*)(dynsm + 128);     // [32][160]
  f16* Hs = A + 32 * CPAD;           // [32][256]
  int c = blockIdx.x, lane = threadIdx.x;
  int p = c / m;
  const int* ni = nidx + (size_t)c * K_;   // uniform base -> scalar loads
  int gi = ni[lane];
  valid[lane] = (gi >= 0);

#ifdef USE_TDM
  // Tensor Data Mover gather: 16 rows of 128 f16 per op; pad 64B after every
  // 256B row so LDS row stride = 160 halves (WMMA A staging layout).
  unsigned ldsA = (unsigned)(size_t)(void*)A;                  // LDS byte offset
  unsigned long long gbase = (unsigned long long)(size_t)(x1h + (size_t)p * M1_ * C);
#pragma unroll
  for (int half = 0; half < 2; ++half) {
    int w[8];
#pragma unroll
    for (int j = 0; j < 8; ++j) {
      int i0 = ni[half * 16 + 2 * j];     i0 = i0 < 0 ? 0 : i0;
      int i1 = ni[half * 16 + 2 * j + 1]; i1 = i1 < 0 ? 0 : i1;
      w[j] = (i0 & 0xffff) | (i1 << 16);
    }
    // D# group0: count=1, gather_mode=1 (16-bit idx), lds_addr, 57b global addr, type=2
    v4u g0 = { 1u | 0x80000000u,
               ldsA + (unsigned)(half * 16 * CPAD * 2),
               (unsigned)gbase,
               (unsigned)((gbase >> 32) & 0x01FFFFFFu) | (2u << 30) };
    // D# group1: data_size=2B, pad_enable, pad_interval=64 DW, pad_amount=16 DW,
    //            dim0=128, dim1=512, tile_dim0=128, tile_dim1=16 idx, stride0=128
    v8i g1 = { (int)0x1F510000, 128 << 16, 512 << 16, 128 << 16, 16, 128, 0, 0 };
    v4i g2 = { w[0], w[1], w[2], w[3] };
    v4i g3 = { w[4], w[5], w[6], w[7] };
#if __clang_major__ >= 23
    v8i gz = { 0, 0, 0, 0, 0, 0, 0, 0 };
    __builtin_amdgcn_tensor_load_to_lds(g0, g1, g2, g3, gz, 0);
#else
    __builtin_amdgcn_tensor_load_to_lds(g0, g1, g2, g3, 0);
#endif
  }
  __builtin_amdgcn_s_wait_tensorcnt(0);
  // zero rows of invalid neighbors (TDM fetched row 0 there)
  f16* Ar = A + lane * CPAD;
  if (gi < 0)
    for (int j = 0; j < C; ++j) Ar[j] = (f16)0.0f;
#else
  int gsf = gi < 0 ? 0 : gi;
  float keepf = gi >= 0 ? 1.0f : 0.0f;
  const f16* F = x1h + ((size_t)p * n + gsf) * C;
  f16* Ar = A + lane * CPAD;
  for (int j = 0; j < C; ++j) Ar[j] = (f16)((float)F[j] * keepf);
#endif
  // relative position cols + zero pad
  int gs = gi < 0 ? 0 : gi;
  float keep = gi >= 0 ? 1.0f : 0.0f;
  float qx = q[(size_t)c * 3], qy = q[(size_t)c * 3 + 1], qz = q[(size_t)c * 3 + 2];
  const float* Pt = pts + ((size_t)p * n + gs) * 3;
  Ar[C + 0] = (f16)((Pt[0] - qx) * keep);
  Ar[C + 1] = (f16)((Pt[1] - qy) * keep);
  Ar[C + 2] = (f16)((Pt[2] - qz) * keep);
  for (int j = C + 3; j < CPAD; ++j) Ar[j] = (f16)0.0f;
  __builtin_prefetch(WaPk, 0, 1);
  __builtin_prefetch(WbPk, 0, 1);
  __syncthreads();

  int cr = lane & 15;
  int kb = (lane >> 4) << 3;
  for (int mt = 0; mt < 2; ++mt)
    for (int jt = 0; jt < H1 / 16; ++jt) {
      v8f acc = {};
      for (int kt = 0; kt < CPAD / 32; ++kt) {
        v16h a = load_a_frag(A + (size_t)mt * 16 * CPAD + kt * 32, CPAD, cr, kb);
        v16h b = load_b_pk(WaPk, H1 / 16, kt, jt, lane);
        acc = __builtin_amdgcn_wmma_f32_16x16x32_f16(false, a, false, b, (short)0, acc, false, false);
      }
      float bc = ba[jt * 16 + cr];
#pragma unroll
      for (int r = 0; r < 8; ++r) {
        int rr = mt * 16 + ((lane < 16) ? r : 8 + r);
        float v = acc[r] + bc; v = v > 0.0f ? v : 0.0f;
        Hs[(size_t)rr * H1 + jt * 16 + cr] = (f16)v;
      }
    }
  __syncthreads();
  for (int jt = 0; jt < H2 / 16; ++jt) {
    float cmax = -INFF;
    float bc = bb[jt * 16 + cr];
    for (int mt = 0; mt < 2; ++mt) {
      v8f acc = {};
      for (int kt = 0; kt < H1 / 32; ++kt) {
        v16h a = load_a_frag(Hs + (size_t)mt * 16 * H1 + kt * 32, H1, cr, kb);
        v16h b = load_b_pk(WbPk, H2 / 16, kt, jt, lane);
        acc = __builtin_amdgcn_wmma_f32_16x16x32_f16(false, a, false, b, (short)0, acc, false, false);
      }
#pragma unroll
      for (int r = 0; r < 8; ++r) {
        int rr = mt * 16 + ((lane < 16) ? r : 8 + r);
        float v = acc[r] + bc; v = v > 0.0f ? v : 0.0f;
        if (valid[rr]) cmax = fmaxf(cmax, v);
      }
    }
    cmax = fmaxf(cmax, __shfl_xor(cmax, 16, 32));
    if (lane < 16) out[(size_t)c * H2 + jt * 16 + cr] = cmax;
  }
}

// ---------------- final MLP + global max pool ----------------
__global__ void k_final(const float* __restrict__ x2, const float* __restrict__ q2,
                        const f16* __restrict__ WaPk, const float* __restrict__ ba,
                        const f16* __restrict__ WbPk, const float* __restrict__ bb,
                        float* __restrict__ xg) {
  const int C = 384, CPAD = 416, H1 = 512, H2 = 768;
  extern __shared__ char dynsm[];
  f16* A  = (f16*)dynsm;
  f16* Hs = A + 32 * CPAD;
  int blk = blockIdx.x, lane = threadIdx.x;
  int p = blk >> 2, rt = blk & 3;
  int row = p * M2_ + rt * 32 + lane;
  const float* F = x2 + (size_t)row * C;
  const float* Q = q2 + (size_t)row * 3;
  f16* Ar = A + lane * CPAD;
  for (int j = 0; j < C; ++j) Ar[j] = (f16)F[j];
  Ar[C + 0] = (f16)Q[0]; Ar[C + 1] = (f16)Q[1]; Ar[C + 2] = (f16)Q[2];
  for (int j = C + 3; j < CPAD; ++j) Ar[j] = (f16)0.0f;
  __builtin_prefetch(WaPk, 0, 1);
  __builtin_prefetch(WbPk, 0, 1);
  __syncthreads();

  int cr = lane & 15;
  int kb = (lane >> 4) << 3;
  for (int mt = 0; mt < 2; ++mt)
    for (int jt = 0; jt < H1 / 16; ++jt) {
      v8f acc = {};
      for (int kt = 0; kt < CPAD / 32; ++kt) {
        v16h a = load_a_frag(A + (size_t)mt * 16 * CPAD + kt * 32, CPAD, cr, kb);
        v16h b = load_b_pk(WaPk, H1 / 16, kt, jt, lane);
        acc = __builtin_amdgcn_wmma_f32_16x16x32_f16(false, a, false, b, (short)0, acc, false, false);
      }
      float bc = ba[jt * 16 + cr];
#pragma unroll
      for (int r = 0; r < 8; ++r) {
        int rr = mt * 16 + ((lane < 16) ? r : 8 + r);
        float v = acc[r] + bc; v = v > 0.0f ? v : 0.0f;
        Hs[(size_t)rr * H1 + jt * 16 + cr] = (f16)v;
      }
    }
  __syncthreads();
  for (int jt = 0; jt < H2 / 16; ++jt) {
    float cmax = 0.0f;
    float bc = bb[jt * 16 + cr];
    for (int mt = 0; mt < 2; ++mt) {
      v8f acc = {};
      for (int kt = 0; kt < H1 / 32; ++kt) {
        v16h a = load_a_frag(Hs + (size_t)mt * 16 * H1 + kt * 32, H1, cr, kb);
        v16h b = load_b_pk(WbPk, H2 / 16, kt, jt, lane);
        acc = __builtin_amdgcn_wmma_f32_16x16x32_f16(false, a, false, b, (short)0, acc, false, false);
      }
#pragma unroll
      for (int r = 0; r < 8; ++r) {
        float v = acc[r] + bc; v = v > 0.0f ? v : 0.0f;
        cmax = fmaxf(cmax, v);
      }
    }
    cmax = fmaxf(cmax, __shfl_xor(cmax, 16, 32));
    if (lane < 16)  // ReLU outputs >= 0: int-bit compare is monotonic
      atomicMax((int*)&xg[(size_t)p * H2 + jt * 16 + cr], __float_as_int(cmax));
  }
}

// ======================= host launch =======================
extern "C" void kernel_launch(void* const* d_in, const int* in_sizes, int n_in,
                              void* d_out, int out_size, void* d_ws, size_t ws_size,
                              hipStream_t stream) {
  (void)in_sizes; (void)n_in; (void)out_size; (void)ws_size;
  const float* pos = (const float*)d_in[0];
  const float* W1a = (const float*)d_in[2];  const float* b1a = (const float*)d_in[3];
  const float* W1b = (const float*)d_in[4];  const float* b1b = (const float*)d_in[5];
  const float* W2a = (const float*)d_in[6];  const float* b2a = (const float*)d_in[7];
  const float* W2b = (const float*)d_in[8];  const float* b2b = (const float*)d_in[9];
  const float* W3a = (const float*)d_in[10]; const float* b3a = (const float*)d_in[11];
  const float* W3b = (const float*)d_in[12]; const float* b3b = (const float*)d_in[13];
  float* out = (float*)d_out;

  char* ws = (char*)d_ws;
  auto carve = [&](size_t bytes) { char* r = ws; ws += (bytes + 255) & ~(size_t)255; return r; };
  float* pn    = (float*)carve(sizeof(float) * P_ * N_ * 3);
  float* q1    = (float*)carve(sizeof(float) * P_ * M1_ * 3);
  int*   idx1  = (int*)  carve(sizeof(int)   * P_ * M1_);
  int*   nidx1 = (int*)  carve(sizeof(int)   * (size_t)P_ * M1_ * K_);
  f16*   x1h   = (f16*)  carve(sizeof(f16)   * (size_t)P_ * M1_ * 128);
  int*   idx2  = (int*)  carve(sizeof(int)   * P_ * M2_);
  int*   nidx2 = (int*)  carve(sizeof(int)   * P_ * M2_ * K_);
  f16* hW1a = (f16*)carve(sizeof(f16) * 32  * 64);
  f16* hW1b = (f16*)carve(sizeof(f16) * 64  * 128);
  f16* hW2a = (f16*)carve(sizeof(f16) * 160 * 256);
  f16* hW2b = (f16*)carve(sizeof(f16) * 256 * 384);
  f16* hW3a = (f16*)carve(sizeof(f16) * 416 * 512);
  f16* hW3b = (f16*)carve(sizeof(f16) * 512 * 768);

  k_aux<<<(P_ * 768 + 255) / 256, 256, 0, stream>>>(out);
  k_cvt_pack<<<(32 * 64 + 255) / 256, 256, 0, stream>>>(W1a, hW1a, 6, 64, 32);
  k_cvt_pack<<<(64 * 128 + 255) / 256, 256, 0, stream>>>(W1b, hW1b, 64, 128, 64);
  k_cvt_pack<<<(160 * 256 + 255) / 256, 256, 0, stream>>>(W2a, hW2a, 131, 256, 160);
  k_cvt_pack<<<(256 * 384 + 255) / 256, 256, 0, stream>>>(W2b, hW2b, 256, 384, 256);
  k_cvt_pack<<<(416 * 512 + 255) / 256, 256, 0, stream>>>(W3a, hW3a, 387, 512, 416);
  k_cvt_pack<<<(512 * 768 + 255) / 256, 256, 0, stream>>>(W3b, hW3b, 512, 768, 512);

  k_norm<<<P_, 256, 0, stream>>>(pos, pn, out + OFF_VMIN, out + OFF_DIFF);

  // layer 1
  k_fps<<<P_, 256, (N_ + 512) * sizeof(float), stream>>>(pn, N_, M1_, idx1, q1);
  k_nbr<<<P_ * M1_ / 8, 256, (size_t)8 * N_ * sizeof(float), stream>>>(pn, q1, N_, M1_, R1_ * R1_, nidx1);
  k_conv<<<P_ * M1_, 32, 128 + sizeof(f16) * (32 * 32 + 32 * 64), stream>>>(
      pn, 3, pn, q1, nidx1, hW1a, b1a, 32, 64, hW1b, b1b, 128, x1h, N_, M1_);

  // layer 2 (TDM gather staging)
  k_fps<<<P_, 256, (M1_ + 512) * sizeof(float), stream>>>(q1, M1_, M2_, idx2, out + OFF_Q2);
  k_nbr<<<P_ * M2_ / 8, 256, (size_t)8 * M1_ * sizeof(float), stream>>>(q1, out + OFF_Q2, M1_, M2_, R2_ * R2_, nidx2);
  k_conv2<<<P_ * M2_, 32, 128 + sizeof(f16) * (32 * 160 + 32 * 256), stream>>>(
      x1h, q1, out + OFF_Q2, nidx2, hW2a, b2a, hW2b, b2b, out + OFF_X2);

  // final
  k_final<<<P_ * 4, 32, sizeof(f16) * (32 * 416 + 32 * 512), stream>>>(
      out + OFF_X2, out + OFF_Q2, hW3a, b3a, hW3b, b3b, out + OFF_XG);
}
